// VSSBlock_16939351015765
// MI455X (gfx1250) — matrix operand order repaired and checked by reading, not compiled, not run on hardware
//
#include <hip/hip_runtime.h>
#include <hip/hip_bf16.h>
#include <cstdint>
#include <cstddef>

// ---------------- problem constants (B,H,W,C) = (2,128,128,96) ----------------
constexpr int Bn   = 2;
constexpr int Hh   = 128;
constexpr int Wcol = 128;
constexpr int Cc   = 96;
constexpr int Dd   = 2 * Cc;          // 192  d_inner
constexpr int Nn   = 16;              // d_state
constexpr int Rr   = 6;               // dt_rank
constexpr int RN2  = Rr + 2 * Nn;     // 38
constexpr int RN2P = 48;              // padded to 3 x 16 WMMA col tiles
constexpr int Ll   = Hh * Wcol;       // 16384
constexpr int Rows = Bn * Ll;         // 32768
constexpr int SGH  = 6 * Nn;          // 96
constexpr int CH   = 16;              // scan chunks
constexpr int CLEN = Ll / CH;         // 1024
constexpr int TST  = 64;              // scan LDS stage length
constexpr int TPAD = TST + 4;         // keeps 16B alignment of rows AND rotates banks

typedef __bf16 bf16_t;
typedef bf16_t v16bf __attribute__((ext_vector_type(16)));
typedef bf16_t v8bf  __attribute__((ext_vector_type(8)));
typedef float  v8f   __attribute__((ext_vector_type(8)));

__device__ inline v8f wmma_bf16(v16bf a, v16bf b, v8f c) {
  // (neg_a, A, neg_b, B, c_mod, C, reuse_a, reuse_b) -> v_wmma_f32_16x16x32_bf16
  return __builtin_amdgcn_wmma_f32_16x16x32_bf16(false, a, false, b, (short)0, c, false, false);
}

__device__ inline v16bf cat8(v8bf lo, v8bf hi) {
  return __builtin_shufflevector(lo, hi, 0,1,2,3,4,5,6,7,8,9,10,11,12,13,14,15);
}

// async memory -> LDS staging (gfx1250 GLOBAL_LOAD_ASYNC_TO_LDS_B128, ASYNCcnt)
__device__ inline void async_lds_b128(void* lds_dst, const void* gsrc) {
  const uint32_t loff = (uint32_t)(uintptr_t)lds_dst;   // flat low 32 bits = LDS offset
  asm volatile("global_load_async_to_lds_b128 %0, %1, off"
               :: "v"(loff), "v"(gsrc) : "memory");
}
__device__ inline void wait_async0() {
  asm volatile("s_wait_asynccnt 0" ::: "memory");
}

// ---------------------------------------------------------------------------
// WMMA GEMM:  out[M x Ncols] = X[M x KDIM](bf16) * W[Ncols x KDIM]^T(bf16)
// Per-wave tile: (MT*16) x (NT*16); K fully unrolled; all fragment loads of a
// k-step are issued before the WMMAs so waits amortize over MT*NT wmma ops.
// A-frag (16-bit 16x32): lanes 0-15 rows, halves K{0..7,16..23}; lanes 16-31
// halves K{8..15,24..31}.  B-frag: col = lane&15, same K order as A-frag.
// MODE 0: f32 row-major out (+opt residual); 1: bf16 row-major; 2: f32 (B,Ncols,Lr).
// ---------------------------------------------------------------------------
template<int MT, int NT, int KDIM, int MODE, bool RESID>
__global__ __launch_bounds__(128) void gemm_wmma_kernel(
    const bf16_t* __restrict__ X, const bf16_t* __restrict__ W,
    void* __restrict__ outp, const float* __restrict__ resid,
    int M, int Ncols, int Lr)
{
  const int lane   = threadIdx.x & 31;
  const int wave   = threadIdx.x >> 5;
  const int mtile  = (blockIdx.x * 4 + wave) * (MT * 16);
  const int ntile0 = blockIdx.y * (NT * 16);
  if (mtile >= M) return;
  const int hi   = lane >> 4;
  const int lmod = lane & 15;

  v8f acc[MT][NT];
#pragma unroll
  for (int mt = 0; mt < MT; ++mt)
#pragma unroll
    for (int t = 0; t < NT; ++t)
#pragma unroll
      for (int j = 0; j < 8; ++j) acc[mt][t][j] = 0.0f;

  const bf16_t* xrow[MT];
#pragma unroll
  for (int mt = 0; mt < MT; ++mt)
    xrow[mt] = X + (size_t)(mtile + mt * 16 + lmod) * KDIM;

#pragma unroll
  for (int k0 = 0; k0 < KDIM; k0 += 32) {
    // --- issue every load of this k-step first ---
    v16bf a[MT];
#pragma unroll
    for (int mt = 0; mt < MT; ++mt) {
      const v8bf alo = *(const v8bf*)(xrow[mt] + k0 + hi * 8);
      const v8bf ahi = *(const v8bf*)(xrow[mt] + k0 + 16 + hi * 8);
      a[mt] = cat8(alo, ahi);
    }
    v16bf bf[NT];
#pragma unroll
    for (int t = 0; t < NT; ++t) {
      const bf16_t* wp = W + (size_t)(ntile0 + t * 16 + lmod) * KDIM + k0 + hi * 8;
      bf[t] = cat8(*(const v8bf*)wp, *(const v8bf*)(wp + 16));
    }
    // --- then the MT*NT matrix ops ---
#pragma unroll
    for (int mt = 0; mt < MT; ++mt)
#pragma unroll
      for (int t = 0; t < NT; ++t)
        acc[mt][t] = wmma_bf16(a[mt], bf[t], acc[mt][t]);
  }

#pragma unroll
  for (int mt = 0; mt < MT; ++mt) {
#pragma unroll
    for (int t = 0; t < NT; ++t) {
      const int col = ntile0 + t * 16 + lmod;
      if (col >= Ncols) continue;
#pragma unroll
      for (int rr = 0; rr < 8; ++rr) {
        const int row = mtile + mt * 16 + rr + hi * 8;  // C/D: vgpr rr, hi-half -> +8
        float v = acc[mt][t][rr];
        if (MODE == 2) {
          const int bb = row / Lr;
          const int ll = row - bb * Lr;
          ((float*)outp)[((size_t)bb * Ncols + col) * Lr + ll] = v;
        } else {
          const size_t idx = (size_t)row * Ncols + col;
          if (RESID) v += resid[idx];
          if (MODE == 1) ((bf16_t*)outp)[idx] = (bf16_t)v;
          else           ((float*)outp)[idx]  = v;
        }
      }
    }
  }
}

// ---------------- f32 -> bf16 convert with row padding (for weights) ----------------
__global__ __launch_bounds__(256) void cvt_pad_bf16_kernel(
    const float* __restrict__ src, bf16_t* __restrict__ dst,
    int rows, int K, int rowsPad)
{
  const long long i = (long long)blockIdx.x * 256 + threadIdx.x;
  if (i >= (long long)rowsPad * K) return;
  const int row = (int)(i / K);
  dst[i] = (row < rows) ? (bf16_t)src[i] : (bf16_t)0.0f;
}

// ---------------- LayerNorm over C=96, one wave32 per row, bf16 out ----------------
__global__ __launch_bounds__(128) void ln_c_kernel(
    const float* __restrict__ x, const float* __restrict__ g,
    const float* __restrict__ b, bf16_t* __restrict__ out)
{
  const int lane = threadIdx.x & 31;
  const int row  = blockIdx.x * 4 + (threadIdx.x >> 5);
  const float* xr = x + (size_t)row * Cc;
  float v[3]; float s = 0.f, ss = 0.f;
#pragma unroll
  for (int i = 0; i < 3; ++i) { v[i] = xr[lane + 32 * i]; s += v[i]; ss += v[i] * v[i]; }
#pragma unroll
  for (int m = 16; m >= 1; m >>= 1) { s += __shfl_xor(s, m, 32); ss += __shfl_xor(ss, m, 32); }
  const float mean = s * (1.0f / Cc);
  const float var  = ss * (1.0f / Cc) - mean * mean;
  const float rstd = rsqrtf(var + 1e-5f);
  bf16_t* orow = out + (size_t)row * Cc;
#pragma unroll
  for (int i = 0; i < 3; ++i) {
    const int c = lane + 32 * i;
    orow[c] = (bf16_t)((v[i] - mean) * rstd * g[c] + b[c]);
  }
}

// ---- depthwise 3x3 conv + bias + SiLU; emits (B,D,L) f32 (scan u) and (B*L,D) bf16 ----
__global__ __launch_bounds__(256) void conv2d_dw_kernel(
    const float* __restrict__ xz, const float* __restrict__ w,
    const float* __restrict__ bias,
    float* __restrict__ xs_f, bf16_t* __restrict__ xs_bf)
{
  const int d  = blockIdx.y * 64 + (threadIdx.x & 63);
  const int pl = blockIdx.x * 4 + (threadIdx.x >> 6);   // b*L + l
  const int b  = pl >> 14, l = pl & (Ll - 1);
  const int h  = l >> 7,   px = l & (Wcol - 1);
  float s = bias[d];
#pragma unroll
  for (int kh = -1; kh <= 1; ++kh) {
    const int h2 = h + kh;
    if (h2 < 0 || h2 >= Hh) continue;
#pragma unroll
    for (int kw = -1; kw <= 1; ++kw) {
      const int w2 = px + kw;
      if (w2 < 0 || w2 >= Wcol) continue;
      s += w[d * 9 + (kh + 1) * 3 + (kw + 1)] *
           xz[((size_t)((b << 14) | (h2 << 7) | w2)) * (2 * Dd) + d];
    }
  }
  const float v = s / (1.0f + expf(-s));                 // SiLU
  xs_f[(((size_t)(b * Dd + d)) << 14) | l] = v;
  xs_bf[(size_t)pl * Dd + d] = (bf16_t)v;
}

// ---------------- SimpleGate1D (GLU MLP over N=16), weights staged in LDS ----------------
__global__ __launch_bounds__(256) void simple_gate_kernel(
    const float* __restrict__ low_dbl,
    const float* __restrict__ w1B, const float* __restrict__ w2B,
    const float* __restrict__ w1C, const float* __restrict__ w2C,
    float* __restrict__ outB, float* __restrict__ outC)
{
  __shared__ float smem[2 * SGH * Nn + Nn * SGH];        // 3072 + 1536 f32
  float* s_w1 = smem;
  float* s_w2 = smem + 2 * SGH * Nn;
  const int gate = blockIdx.y;
  const float* w1 = gate ? w1C : w1B;
  const float* w2 = gate ? w2C : w2B;
  for (int i = threadIdx.x; i < 2 * SGH * Nn + Nn * SGH; i += 256) {
    if (i < 2 * SGH * Nn) s_w1[i] = w1[i];
    else                  s_w2[i - 2 * SGH * Nn] = w2[i - 2 * SGH * Nn];
  }
  __syncthreads();
  const int pl = blockIdx.x * 256 + threadIdx.x;
  const int b = pl >> 14, l = pl & (Ll - 1);
  const float* src = low_dbl + ((size_t)(b * RN2 + Rr + gate * Nn)) * Ll + l;
  float v[Nn], o[Nn];
#pragma unroll
  for (int n = 0; n < Nn; ++n) { v[n] = src[(size_t)n * Ll]; o[n] = 0.f; }
  for (int j = 0; j < SGH; ++j) {
    float h1 = 0.f, h2 = 0.f;
#pragma unroll
    for (int n = 0; n < Nn; ++n) {
      h1 += s_w1[j * Nn + n] * v[n];
      h2 += s_w1[(j + SGH) * Nn + n] * v[n];
    }
    const float ge = 0.5f * h1 * (1.0f + erff(h1 * 0.70710678118654752f)); // exact gelu
    const float gg = ge * h2;
#pragma unroll
    for (int c = 0; c < Nn; ++c) o[c] += s_w2[c * SGH + j] * gg;
  }
  float* dst = gate ? outC : outB;
#pragma unroll
  for (int c = 0; c < Nn; ++c) dst[((size_t)(b * Nn + c)) * Ll + l] = o[c];
}

// ------------- dilated (rate 2) depthwise conv1d k=7 pad=6 over L, with low-branch add -------------
__global__ __launch_bounds__(256) void dwconv1d_kernel(
    const float* __restrict__ x_dbl,
    const float* __restrict__ bsg, const float* __restrict__ csg,
    const float* __restrict__ wdt, const float* __restrict__ wB, const float* __restrict__ wC,
    float* __restrict__ dtsc, float* __restrict__ bsc, float* __restrict__ csc)
{
  const int idx = blockIdx.x * 256 + threadIdx.x;        // over B*38*L
  const int l   = idx & (Ll - 1);
  const int c38 = (idx >> 14) % RN2;
  const int b   = idx / (RN2 * Ll);
  const float* w; const float* add = nullptr; float* out;
  if (c38 < Rr) {
    w = wdt + c38 * 7;                        out = dtsc + ((size_t)(b * Rr + c38)) * Ll;
  } else if (c38 < Rr + Nn) {
    const int n = c38 - Rr;
    w = wB + n * 7; add = bsg + ((size_t)(b * Nn + n)) * Ll;
    out = bsc + ((size_t)(b * Nn + n)) * Ll;
  } else {
    const int n = c38 - Rr - Nn;
    w = wC + n * 7; add = csg + ((size_t)(b * Nn + n)) * Ll;
    out = csc + ((size_t)(b * Nn + n)) * Ll;
  }
  const float* in = x_dbl + ((size_t)(b * RN2 + c38)) * Ll;
  float s = 0.f;
#pragma unroll
  for (int i = 0; i < 7; ++i) {
    const int t = l + 2 * i - 6;
    if (t >= 0 && t < Ll) {
      float xv = in[t];
      if (add) xv += add[t];
      s += w[i] * xv;
    }
  }
  out[l] = s;
}

// ---------------- dt_proj (K=6) + softplus -> delta (B,D,L) ----------------
__global__ __launch_bounds__(256) void delta_kernel(
    const float* __restrict__ dtsc, const float* __restrict__ dtw,
    const float* __restrict__ dtb, float* __restrict__ delta)
{
  const int idx = blockIdx.x * 256 + threadIdx.x;        // over B*D*L
  const int l = idx & (Ll - 1);
  const int d = (idx >> 14) % Dd;
  const int b = idx / (Dd * Ll);
  float s = dtb[d];
#pragma unroll
  for (int r = 0; r < Rr; ++r)
    s += dtw[d * Rr + r] * dtsc[((size_t)(b * Rr + r)) * Ll + l];
  delta[idx] = (s > 20.f) ? s : log1pf(expf(s));
}

// ---------------- selective scan, pass 1: per-chunk local state + sum(dt) ----------------
// block = 256 thr = 16 d-groups x 16 state lanes; grid = (24 bd-tiles, 16 chunks)
// tiles staged memory->LDS with async B128 ops (ASYNCcnt) instead of VGPR round-trips
__global__ __launch_bounds__(256) void scan_chunk_kernel(
    const float* __restrict__ delta, const float* __restrict__ u,
    const float* __restrict__ bsc, const float* __restrict__ A_logs,
    float* __restrict__ Sout, float* __restrict__ sumdt)
{
  __shared__ __align__(16) float s_b[16][TPAD], s_dt[16][TPAD], s_u[16][TPAD];
  const int bi = blockIdx.x;
  const int b = bi / 12, dbase = (bi % 12) * 16;
  const int chunk = blockIdx.y;
  const int g = threadIdx.x >> 4, n = threadIdx.x & 15;
  const int d = dbase + g;
  const float An = -expf(A_logs[d * Nn + n]);
  float h = 0.f, sdt = 0.f;
  const int t0 = chunk * CLEN;
  const int rw = threadIdx.x >> 4;      // staging row 0..15
  const int ch = (threadIdx.x & 15) * 4; // 16B chunk within the 64-wide tile
  for (int tt = 0; tt < CLEN; tt += TST) {
    async_lds_b128(&s_b[rw][ch],  bsc   + (((size_t)(b * Nn + rw)) << 14) + t0 + tt + ch);
    async_lds_b128(&s_dt[rw][ch], delta + (((size_t)(b * Dd + dbase + rw)) << 14) + t0 + tt + ch);
    async_lds_b128(&s_u[rw][ch],  u     + (((size_t)(b * Dd + dbase + rw)) << 14) + t0 + tt + ch);
    wait_async0();
    __syncthreads();
    for (int j = 0; j < TST; ++j) {
      const float dt = s_dt[g][j];
      h = expf(dt * An) * h + dt * s_u[g][j] * s_b[n][j];
      sdt += dt;
    }
    __syncthreads();
  }
  Sout[(((size_t)(b * Dd + d)) * CH + chunk) * Nn + n] = h;
  if (n == 0) sumdt[((size_t)(b * Dd + d)) * CH + chunk] = sdt;
}

// ---------------- pass 2: prefix over the 16 chunks (chunk transition = exp(A*sum_dt)) ----------------
__global__ __launch_bounds__(256) void scan_fix_kernel(
    const float* __restrict__ Sin, const float* __restrict__ sumdt,
    const float* __restrict__ A_logs, float* __restrict__ hstart)
{
  const int tid = blockIdx.x * 256 + threadIdx.x;        // B*D*N = 6144
  const int n  = tid & 15;
  const int bd = tid >> 4;
  const int d  = bd % Dd;
  const float An = -expf(A_logs[d * Nn + n]);
  float h = 0.f;
  for (int c = 0; c < CH; ++c) {
    const size_t idx = ((size_t)bd * CH + c) * Nn + n;
    hstart[idx] = h;
    h = expf(An * sumdt[(size_t)bd * CH + c]) * h + Sin[idx];
  }
}

// ---------------- pass 3: replay with correct h0, emit y = h.C + Ds*u  (B,L,D) ----------------
__global__ __launch_bounds__(256) void scan_apply_kernel(
    const float* __restrict__ delta, const float* __restrict__ u,
    const float* __restrict__ bsc, const float* __restrict__ csc,
    const float* __restrict__ A_logs, const float* __restrict__ Ds,
    const float* __restrict__ hstart, float* __restrict__ yout)
{
  __shared__ __align__(16) float s_b[16][TPAD], s_c[16][TPAD], s_dt[16][TPAD], s_u[16][TPAD];
  const int bi = blockIdx.x;
  const int b = bi / 12, dbase = (bi % 12) * 16;
  const int chunk = blockIdx.y;
  const int g = threadIdx.x >> 4, n = threadIdx.x & 15;
  const int d = dbase + g;
  const float An  = -expf(A_logs[d * Nn + n]);
  const float Dsd = Ds[d];
  float h = hstart[(((size_t)(b * Dd + d)) * CH + chunk) * Nn + n];
  const int t0 = chunk * CLEN;
  const int rw = threadIdx.x >> 4;
  const int ch = (threadIdx.x & 15) * 4;
  for (int tt = 0; tt < CLEN; tt += TST) {
    async_lds_b128(&s_b[rw][ch],  bsc   + (((size_t)(b * Nn + rw)) << 14) + t0 + tt + ch);
    async_lds_b128(&s_c[rw][ch],  csc   + (((size_t)(b * Nn + rw)) << 14) + t0 + tt + ch);
    async_lds_b128(&s_dt[rw][ch], delta + (((size_t)(b * Dd + dbase + rw)) << 14) + t0 + tt + ch);
    async_lds_b128(&s_u[rw][ch],  u     + (((size_t)(b * Dd + dbase + rw)) << 14) + t0 + tt + ch);
    wait_async0();
    __syncthreads();
    for (int j = 0; j < TST; ++j) {
      const float dt = s_dt[g][j];
      const float uu = s_u[g][j];
      h = expf(dt * An) * h + dt * uu * s_b[n][j];
      float y = h * s_c[n][j];
      y += __shfl_xor(y, 1, 16);
      y += __shfl_xor(y, 2, 16);
      y += __shfl_xor(y, 4, 16);
      y += __shfl_xor(y, 8, 16);
      if (n == 0)
        yout[((size_t)((b << 14) + t0 + tt + j)) * Dd + d] = y + Dsd * uu;
    }
    __syncthreads();
  }
}

// ---------------- out-LayerNorm over D=192 + silu(z) gate, bf16 out ----------------
__global__ __launch_bounds__(128) void ln_gate_kernel(
    const float* __restrict__ y, const float* __restrict__ xz,
    const float* __restrict__ g, const float* __restrict__ b,
    bf16_t* __restrict__ out)
{
  const int lane = threadIdx.x & 31;
  const int row  = blockIdx.x * 4 + (threadIdx.x >> 5);
  const float* yr = y + (size_t)row * Dd;
  float v[6]; float s = 0.f, ss = 0.f;
#pragma unroll
  for (int i = 0; i < 6; ++i) { v[i] = yr[lane + 32 * i]; s += v[i]; ss += v[i] * v[i]; }
#pragma unroll
  for (int m = 16; m >= 1; m >>= 1) { s += __shfl_xor(s, m, 32); ss += __shfl_xor(ss, m, 32); }
  const float mean = s * (1.0f / Dd);
  const float var  = ss * (1.0f / Dd) - mean * mean;
  const float rstd = rsqrtf(var + 1e-5f);
  const float* zr = xz + (size_t)row * (2 * Dd) + Dd;
  bf16_t* orow = out + (size_t)row * Dd;
#pragma unroll
  for (int i = 0; i < 6; ++i) {
    const int c = lane + 32 * i;
    const float z  = zr[c];
    const float sz = z / (1.0f + expf(-z));
    orow[c] = (bf16_t)(((v[i] - mean) * rstd * g[c] + b[c]) * sz);
  }
}

// ===========================================================================
extern "C" void kernel_launch(void* const* d_in, const int* in_sizes, int n_in,
                              void* d_out, int out_size, void* d_ws, size_t ws_size,
                              hipStream_t stream) {
  (void)in_sizes; (void)n_in; (void)out_size; (void)ws_size;
  const float* x             = (const float*)d_in[0];
  const float* hbl           = (const float*)d_in[1];
  const float* ln_g          = (const float*)d_in[2];
  const float* ln_b          = (const float*)d_in[3];
  const float* in_proj_w     = (const float*)d_in[4];
  const float* in_proj_low_w = (const float*)d_in[5];
  const float* conv2d_w      = (const float*)d_in[6];
  const float* conv2d_b      = (const float*)d_in[7];
  const float* x_proj_w      = (const float*)d_in[8];
  const float* x_proj_w_low  = (const float*)d_in[9];
  const float* conv_dt_w     = (const float*)d_in[10];
  const float* conv_B_w      = (const float*)d_in[11];
  const float* conv_C_w      = (const float*)d_in[12];
  const float* sgb_w1        = (const float*)d_in[13];
  const float* sgb_w2        = (const float*)d_in[14];
  const float* sgc_w1        = (const float*)d_in[15];
  const float* sgc_w2        = (const float*)d_in[16];
  const float* dt_proj_w     = (const float*)d_in[17];
  const float* dt_proj_b     = (const float*)d_in[18];
  const float* A_logs        = (const float*)d_in[19];
  const float* Ds            = (const float*)d_in[20];
  const float* outn_g        = (const float*)d_in[21];
  const float* outn_b        = (const float*)d_in[22];
  const float* out_proj_w    = (const float*)d_in[23];

  char* p = (char*)d_ws;
  auto carve = [&](size_t bytes) -> char* {
    char* q = p; p += (bytes + 255) & ~(size_t)255; return q;
  };

  bf16_t* w_in    = (bf16_t*)carve((size_t)(2 * Dd) * Cc * 2);
  bf16_t* w_inlow = (bf16_t*)carve((size_t)Dd * Cc * 2);
  bf16_t* w_xp    = (bf16_t*)carve((size_t)RN2P * Dd * 2);
  bf16_t* w_xpl   = (bf16_t*)carve((size_t)RN2P * Dd * 2);
  bf16_t* w_op    = (bf16_t*)carve((size_t)Cc * Dd * 2);
  bf16_t* hbl_bf  = (bf16_t*)carve((size_t)Rows * Cc * 2);
  bf16_t* xn_bf   = (bf16_t*)carve((size_t)Rows * Cc * 2);
  float*  xz      = (float*) carve((size_t)Rows * 2 * Dd * 4);
  bf16_t* low_bf  = (bf16_t*)carve((size_t)Rows * Dd * 2);
  float*  xs_f    = (float*) carve((size_t)Bn * Dd * Ll * 4);
  bf16_t* xs_bf   = (bf16_t*)carve((size_t)Rows * Dd * 2);
  float*  x_dbl   = (float*) carve((size_t)Bn * RN2 * Ll * 4);
  float*  low_dbl = (float*) carve((size_t)Bn * RN2 * Ll * 4);
  float*  bsg     = (float*) carve((size_t)Bn * Nn * Ll * 4);
  float*  csg     = (float*) carve((size_t)Bn * Nn * Ll * 4);
  float*  dtsc    = (float*) carve((size_t)Bn * Rr * Ll * 4);
  float*  bsc     = (float*) carve((size_t)Bn * Nn * Ll * 4);
  float*  csc     = (float*) carve((size_t)Bn * Nn * Ll * 4);
  float*  delta   = (float*) carve((size_t)Bn * Dd * Ll * 4);
  float*  Sbuf    = (float*) carve((size_t)Bn * Dd * CH * Nn * 4);
  float*  sumdt   = (float*) carve((size_t)Bn * Dd * CH * 4);
  float*  hstart  = (float*) carve((size_t)Bn * Dd * CH * Nn * 4);
  float*  yout    = (float*) carve((size_t)Rows * Dd * 4);
  bf16_t* yg_bf   = (bf16_t*)carve((size_t)Rows * Dd * 2);

  auto cvt = [&](const float* src, bf16_t* dst, int rows, int K, int rowsPad) {
    const long long ncnt = (long long)rowsPad * K;
    cvt_pad_bf16_kernel<<<(unsigned)((ncnt + 255) / 256), 256, 0, stream>>>(src, dst, rows, K, rowsPad);
  };

  // weights / inputs -> bf16
  cvt(in_proj_w,     w_in,    2 * Dd, Cc, 2 * Dd);
  cvt(in_proj_low_w, w_inlow, Dd,     Cc, Dd);
  cvt(x_proj_w,      w_xp,    RN2,    Dd, RN2P);
  cvt(x_proj_w_low,  w_xpl,   RN2,    Dd, RN2P);
  cvt(out_proj_w,    w_op,    Cc,     Dd, Cc);
  cvt(hbl,           hbl_bf,  Rows,   Cc, Rows);

  // ln_1 + in_proj + in_proj_low    (per-wave tile 32x64; K unrolled at compile time)
  ln_c_kernel<<<Rows / 4, 128, 0, stream>>>(x, ln_g, ln_b, xn_bf);
  gemm_wmma_kernel<2, 4, Cc, 0, false><<<dim3(Rows / 128, (2 * Dd) / 64), 128, 0, stream>>>(
      xn_bf, w_in, xz, nullptr, Rows, 2 * Dd, Ll);
  gemm_wmma_kernel<2, 4, Cc, 1, false><<<dim3(Rows / 128, Dd / 64), 128, 0, stream>>>(
      hbl_bf, w_inlow, low_bf, nullptr, Rows, Dd, Ll);

  // depthwise conv2d 3x3 + SiLU
  conv2d_dw_kernel<<<dim3(Rows / 4, Dd / 64), 256, 0, stream>>>(xz, conv2d_w, conv2d_b, xs_f, xs_bf);

  // x_proj / x_proj_low (channel-major out for the 1D convs), K=192
  gemm_wmma_kernel<2, 3, Dd, 2, false><<<dim3(Rows / 128, 1), 128, 0, stream>>>(
      xs_bf, w_xp, x_dbl, nullptr, Rows, RN2, Ll);
  gemm_wmma_kernel<2, 3, Dd, 2, false><<<dim3(Rows / 128, 1), 128, 0, stream>>>(
      low_bf, w_xpl, low_dbl, nullptr, Rows, RN2, Ll);

  // SimpleGate1D on Bs_low / Cs_low, then dilated dw-conv1d, then dt_proj+softplus
  simple_gate_kernel<<<dim3(Rows / 256, 2), 256, 0, stream>>>(
      low_dbl, sgb_w1, sgb_w2, sgc_w1, sgc_w2, bsg, csg);
  dwconv1d_kernel<<<(Bn * RN2 * Ll) / 256, 256, 0, stream>>>(
      x_dbl, bsg, csg, conv_dt_w, conv_B_w, conv_C_w, dtsc, bsc, csc);
  delta_kernel<<<(Bn * Dd * Ll) / 256, 256, 0, stream>>>(dtsc, dt_proj_w, dt_proj_b, delta);

  // chunked selective scan (3 passes), tiles staged via async LDS loads
  scan_chunk_kernel<<<dim3(24, CH), 256, 0, stream>>>(delta, xs_f, bsc, A_logs, Sbuf, sumdt);
  scan_fix_kernel<<<(Bn * Dd * Nn) / 256, 256, 0, stream>>>(Sbuf, sumdt, A_logs, hstart);
  scan_apply_kernel<<<dim3(24, CH), 256, 0, stream>>>(delta, xs_f, bsc, csc, A_logs, Ds, hstart, yout);

  // out-LN * silu(z), then out_proj with fused residual
  ln_gate_kernel<<<Rows / 4, 128, 0, stream>>>(yout, xz, outn_g, outn_b, yg_bf);
  gemm_wmma_kernel<2, 3, Dd, 0, true><<<dim3(Rows / 128, 2), 128, 0, stream>>>(
      yg_bf, w_op, d_out, x, Rows, Cc, Ll);
}
